// VQCodebook_41360535061121
// MI455X (gfx1250) — matrix operand order repaired and checked by reading, not compiled, not run on hardware
//
#include <hip/hip_runtime.h>
#include <hip/hip_bf16.h>

typedef __attribute__((ext_vector_type(16))) _Float16 v16h;
typedef __attribute__((ext_vector_type(8)))  float    v8f;

#define N_TOKENS  262144
#define NUM_CODES 512
#define CODE_DIM  64

// d_ws layout:
//   [0,       65536) : embeddings converted to f16, row-major (512 x 64)
//   [65536,   67584) : e2[n] = ||E[n]||^2  (512 floats)
//   [67584,   67588) : commitment-loss accumulator (1 float)

// ---------------------------------------------------------------------------
// Prep: E f32 -> f16, squared norms, zero loss accumulator.
// ---------------------------------------------------------------------------
__global__ void vq_prep(const float* __restrict__ E,
                        _Float16* __restrict__ e16,
                        float* __restrict__ e2,
                        float* __restrict__ lossAcc) {
    int n = blockIdx.x * blockDim.x + threadIdx.x;
    if (n == 0) *lossAcc = 0.0f;
    if (n < NUM_CODES) {
        const float* row = E + (size_t)n * CODE_DIM;
        _Float16* o = e16 + (size_t)n * CODE_DIM;
        float s = 0.0f;
#pragma unroll
        for (int d = 0; d < CODE_DIM; ++d) {
            float v = row[d];
            s += v * v;
            o[d] = (_Float16)v;
        }
        e2[n] = s;
    }
}

// ---------------------------------------------------------------------------
// Main: per wave, 16 tokens vs all 512 codes via v_wmma_f32_16x16x32_f16.
// Two column tiles in flight per iteration (independent accumulators) to
// fill the WMMA->VALU hazard slots and amortize loop overhead.
// ---------------------------------------------------------------------------
__global__ __launch_bounds__(256) void vq_main(
    const float*    __restrict__ z,      // (N, 64) f32
    const float*    __restrict__ Ef32,   // (512, 64) f32
    const _Float16* __restrict__ e16,    // (512, 64) f16
    const float*    __restrict__ e2,     // (512,)  ||e||^2
    float* __restrict__ zq_out,          // (N, 64) f32
    float* __restrict__ idx_out,         // (N,)    indices as float
    float* __restrict__ lossAcc)         // scalar accumulator in ws
{
    const int lane    = threadIdx.x & 31;
    const int wave    = threadIdx.x >> 5;
    const int rowbase = blockIdx.x * 128 + wave * 16;
    const int col     = lane & 15;   // column / row-within-tile selector
    const int halfsel = lane >> 4;   // which half of the wave

    // ---- A fragments: 16 tokens x 64 dims, f32 -> f16.
    // A layout used here (contiguous k, not the ISA table order): lane (mod 16) = row M; lanes 0-15 hold
    // K = 0..15, lanes 16-31 hold K = 16..31 per 32-K step (matches B).
    const int m = rowbase + col;
    v16h a[2];
#pragma unroll
    for (int ks = 0; ks < 2; ++ks) {
        const int dbase = ks * 32 + halfsel * 16;
        const float4* p0 = (const float4*)(z + (size_t)m * CODE_DIM + dbase);
        const float4* p1 = (const float4*)(z + (size_t)m * CODE_DIM + dbase + 8);
        float4 f0 = p0[0], f1 = p0[1];
        float4 g0 = p1[0], g1 = p1[1];
        v16h av;
        av[0]  = (_Float16)f0.x; av[1]  = (_Float16)f0.y;
        av[2]  = (_Float16)f0.z; av[3]  = (_Float16)f0.w;
        av[4]  = (_Float16)f1.x; av[5]  = (_Float16)f1.y;
        av[6]  = (_Float16)f1.z; av[7]  = (_Float16)f1.w;
        av[8]  = (_Float16)g0.x; av[9]  = (_Float16)g0.y;
        av[10] = (_Float16)g0.z; av[11] = (_Float16)g0.w;
        av[12] = (_Float16)g1.x; av[13] = (_Float16)g1.y;
        av[14] = (_Float16)g1.z; av[15] = (_Float16)g1.w;
        a[ks] = av;
    }

    float best[8];
    int   bidx[8];
#pragma unroll
    for (int r = 0; r < 8; ++r) { best[r] = 3.4e38f; bidx[r] = 0; }

    // ---- sweep 32 column tiles of 16 codes, 2 tiles per iteration.
    for (int t = 0; t < 32; t += 2) {
        const int n0 = t * 16 + col;
        const int n1 = n0 + 16;
        // B layout used here (K x N = 32x16, same k order as A): lane = column N; lanes 0-15 hold K=0..15,
        // lanes 16-31 hold K=16..31, contiguous => one 32B load per k-step.
        const _Float16* bp0 = e16 + (size_t)n0 * CODE_DIM + halfsel * 16;
        const _Float16* bp1 = e16 + (size_t)n1 * CODE_DIM + halfsel * 16;
        v16h b00 = *(const v16h*)(bp0);        // tile t,   d = 0..31 slice
        v16h b01 = *(const v16h*)(bp0 + 32);   // tile t,   d = 32..63 slice
        v16h b10 = *(const v16h*)(bp1);        // tile t+1, d = 0..31 slice
        v16h b11 = *(const v16h*)(bp1 + 32);   // tile t+1, d = 32..63 slice
        const float e2n0 = e2[n0];
        const float e2n1 = e2[n1];

        v8f c0 = {0.f, 0.f, 0.f, 0.f, 0.f, 0.f, 0.f, 0.f};
        v8f c1 = {0.f, 0.f, 0.f, 0.f, 0.f, 0.f, 0.f, 0.f};
        c0 = __builtin_amdgcn_wmma_f32_16x16x32_f16(false, a[0], false, b00,
                                                    (short)0, c0, false, false);
        c1 = __builtin_amdgcn_wmma_f32_16x16x32_f16(false, a[0], false, b10,
                                                    (short)0, c1, false, false);
        c0 = __builtin_amdgcn_wmma_f32_16x16x32_f16(false, a[1], false, b01,
                                                    (short)0, c0, false, false);
        c1 = __builtin_amdgcn_wmma_f32_16x16x32_f16(false, a[1], false, b11,
                                                    (short)0, c1, false, false);

        // C layout: VGPR r holds row (halfsel*8 + r), lane%16 = column.
        // Update in ascending-n order so strict '<' keeps the first min.
#pragma unroll
        for (int r = 0; r < 8; ++r) {
            float v = e2n0 - 2.0f * c0[r];   // ||z||^2 term constant per row
            if (v < best[r]) { best[r] = v; bidx[r] = n0; }
        }
#pragma unroll
        for (int r = 0; r < 8; ++r) {
            float v = e2n1 - 2.0f * c1[r];
            if (v < best[r]) { best[r] = v; bidx[r] = n1; }
        }
    }

    // ---- argmin reduction across the 16 column lanes of each half-wave,
    // first-index tie-break to match jnp.argmin.
#pragma unroll
    for (int r = 0; r < 8; ++r) {
#pragma unroll
        for (int off = 8; off >= 1; off >>= 1) {
            float ov = __shfl_xor(best[r], off, 32);
            int   oi = __shfl_xor(bidx[r], off, 32);
            if (ov < best[r] || (ov == best[r] && oi < bidx[r])) {
                best[r] = ov; bidx[r] = oi;
            }
        }
    }

    // ---- write indices (rows 0-7 from lane 0, rows 8-15 from lane 16)
    if (col == 0) {
#pragma unroll
        for (int r = 0; r < 8; ++r)
            idx_out[rowbase + halfsel * 8 + r] = (float)bidx[r];
    }

    // ---- exact f32 gather of winning rows + commitment-loss partial sum.
    // Whole wave cooperates: each lane moves 2 floats per row (32 lanes * 2 = 64).
    float lsum = 0.0f;
#pragma unroll
    for (int rr = 0; rr < 16; ++rr) {
        int code = __shfl(bidx[rr & 7], (rr < 8) ? 0 : 16, 32);
        const int row = rowbase + rr;
        float2 zq = *(const float2*)(Ef32 + (size_t)code * CODE_DIM + lane * 2);
        float2 zv = *(const float2*)(z    + (size_t)row  * CODE_DIM + lane * 2);
        *(float2*)(zq_out + (size_t)row * CODE_DIM + lane * 2) = zq;
        float d0 = zv.x - zq.x, d1 = zv.y - zq.y;
        lsum += d0 * d0 + d1 * d1;
    }
#pragma unroll
    for (int off = 16; off >= 1; off >>= 1)
        lsum += __shfl_xor(lsum, off, 32);
    if (lane == 0) atomicAdd(lossAcc, lsum);
}

// ---------------------------------------------------------------------------
// Finalize: mean of squared error.
// ---------------------------------------------------------------------------
__global__ void vq_finalize(const float* __restrict__ lossAcc,
                            float* __restrict__ out_loss) {
    *out_loss = *lossAcc * (1.0f / ((float)N_TOKENS * (float)CODE_DIM));
}

extern "C" void kernel_launch(void* const* d_in, const int* in_sizes, int n_in,
                              void* d_out, int out_size, void* d_ws, size_t ws_size,
                              hipStream_t stream) {
    (void)in_sizes; (void)n_in; (void)out_size; (void)ws_size;
    const float* z = (const float*)d_in[0];   // (262144, 64) f32
    const float* E = (const float*)d_in[1];   // (512, 64)    f32

    float* out      = (float*)d_out;
    float* zq_out   = out;                                   // N*64 floats
    float* idx_out  = out + (size_t)N_TOKENS * CODE_DIM;     // N floats
    float* loss_out = idx_out + N_TOKENS;                    // 1 float

    _Float16* e16   = (_Float16*)d_ws;
    float*    e2    = (float*)((char*)d_ws + 65536);
    float*    lacc  = e2 + NUM_CODES;

    vq_prep<<<(NUM_CODES + 255) / 256, 256, 0, stream>>>(E, e16, e2, lacc);
    vq_main<<<N_TOKENS / 128, 256, 0, stream>>>(z, E, e16, e2,
                                                zq_out, idx_out, lacc);
    vq_finalize<<<1, 1, 0, stream>>>(lacc, loss_out);
}
